// BiLSTMDualPathway_40596030881793
// MI455X (gfx1250) — compile-verified, matrix operand order checked by reading, not run on hardware
//
#include <hip/hip_runtime.h>

// ---------------------------------------------------------------------------
// BiLSTM dual pathway for gfx1250 (MI455X).
// Matmuls: v_wmma_f32_16x16x32_bf16 (bf16 in, f32 accumulate).
// GEMM A-tiles staged to LDS via the Tensor Data Mover (tensor_load_to_lds,
// TENSORcnt), double-buffered; falls back to cooperative copies if the
// builtin is unavailable. WMMA operand-reuse hints (RA/RB) on shared operands.
// ---------------------------------------------------------------------------

typedef unsigned short u16t;
typedef __attribute__((ext_vector_type(16))) __bf16 v16bf;
typedef __attribute__((ext_vector_type(8)))  float  v8f;
typedef __attribute__((ext_vector_type(4)))  unsigned int u32x4;
typedef __attribute__((ext_vector_type(8)))  int  i32x8;
typedef __attribute__((ext_vector_type(4)))  int  i32x4;

#if defined(__HIP_DEVICE_COMPILE__) && __has_builtin(__builtin_amdgcn_tensor_load_to_lds)
#define BILSTM_TDM 1
#endif
#if defined(__HIP_DEVICE_COMPILE__) && __has_builtin(__builtin_amdgcn_s_wait_tensorcnt)
#define BILSTM_WAIT_TENSOR(n) __builtin_amdgcn_s_wait_tensorcnt(n)
#else
#define BILSTM_WAIT_TENSOR(n)
#endif

__device__ __forceinline__ float bf2f(u16t u) {
  union { unsigned ui; float fl; } c; c.ui = ((unsigned)u) << 16; return c.fl;
}
__device__ __forceinline__ u16t f2bf(float x) {
  union { float fl; unsigned ui; } c; c.fl = x;
  unsigned r = c.ui + 0x7FFFu + ((c.ui >> 16) & 1u);   // round-to-nearest-even
  return (u16t)(r >> 16);
}
__device__ __forceinline__ __bf16 u2b(u16t u) {
  union { u16t s; __bf16 b; } c; c.s = u; return c.b;
}

__device__ __forceinline__ v8f wmma_bf16(v16bf a, v16bf b, v8f c) {
  return __builtin_amdgcn_wmma_f32_16x16x32_bf16(false, a, false, b,
                                                 (short)0, c, false, false);
}
// reuse-A hint: current instr identical to previous, A operand cached.
__device__ __forceinline__ v8f wmma_bf16_ra(v16bf a, v16bf b, v8f c) {
  return __builtin_amdgcn_wmma_f32_16x16x32_bf16(false, a, false, b,
                                                 (short)0, c, true, false);
}
// reuse-B hint.
__device__ __forceinline__ v8f wmma_bf16_rb(v16bf a, v16bf b, v8f c) {
  return __builtin_amdgcn_wmma_f32_16x16x32_bf16(false, a, false, b,
                                                 (short)0, c, false, true);
}

// Unchecked 16x32 bf16 fragment (global or LDS source). Per CDNA5 7.12.2 each
// lane holds two contiguous 8-element (16B) runs of K.
__device__ __forceinline__ v16bf load_frag_nc(const u16t* base, int ld,
                                              int r0, int k0, int lane) {
  int r  = r0 + (lane & 15);
  int ko = k0 + ((lane >> 4) << 3);
  const u16t* p = base + (long long)r * ld + ko;
  v16bf f;
#pragma unroll
  for (int i = 0; i < 8; ++i) { f[i] = u2b(p[i]); f[i + 8] = u2b(p[i + 16]); }
  return f;
}

// Bounded variant (edge / ragged-N kernel only).
__device__ __forceinline__ v16bf load_frag_b(const u16t* __restrict__ base,
                                             int ld, int r0, int k0,
                                             int lane, int rmax) {
  int r  = r0 + (lane & 15);
  int ko = k0 + ((lane >> 4) << 3);
  v16bf f;
  if (r < rmax) {
    const u16t* p = base + (long long)r * ld + ko;
#pragma unroll
    for (int i = 0; i < 8; ++i) { f[i] = u2b(p[i]); f[i + 8] = u2b(p[i + 16]); }
  } else {
#pragma unroll
    for (int i = 0; i < 16; ++i) f[i] = u2b((u16t)0);
  }
  return f;
}

// Stage a 64(rows) x 32(K) bf16 A-tile into LDS. TDM path: one
// tensor_load_to_lds per block (EXEC ignored for tensor ops); 2D descriptor:
// data_size=2B, tile 32x64, tensor dims (K, M), dim0 stride K, type=2.
// The asm memory clobber makes the DMA's LDS writes visible to the compiler
// so fragment loads from s_a are not folded away.
__device__ __forceinline__ void bilstm_stage_a(
    const u16t* __restrict__ A, int M, int K, int mblk, int k,
    u16t* sa_base, unsigned lds_byte_off, int tid) {
#if BILSTM_TDM
  if (tid == 0) {
    unsigned long long ga =
        (unsigned long long)(size_t)(const void*)(A + (long long)mblk * K + k);
    u32x4 g0;
    g0[0] = 1u;                                  // count=1 valid descriptor
    g0[1] = lds_byte_off;                        // lds_addr
    g0[2] = (unsigned)ga;                        // global_addr[31:0]
    g0[3] = (unsigned)((ga >> 32) & 0x01FFFFFFull) | (2u << 30);  // addr hi | type=2
    i32x8 g1;
    g1[0] = (int)(1u << 16);                                   // data_size=1 (2B)
    g1[1] = (int)(((unsigned)K & 0xFFFFu) << 16);              // tensor_dim0 lo16
    g1[2] = (int)((((unsigned)K >> 16) & 0xFFFFu) |
                  (((unsigned)M & 0xFFFFu) << 16));            // dim0 hi | dim1 lo
    g1[3] = (int)((((unsigned)M >> 16) & 0xFFFFu) | (32u << 16)); // dim1 hi | tile_dim0=32
    g1[4] = 64;                                                // tile_dim1=64
    g1[5] = K;                                                 // dim0_stride lo32
    g1[6] = 0;
    g1[7] = 0;
    i32x4 z4 = {0, 0, 0, 0};
#if __clang_major__ >= 23
    i32x8 z8 = {0, 0, 0, 0, 0, 0, 0, 0};
    __builtin_amdgcn_tensor_load_to_lds(g0, g1, z4, z4, z8, 0);
#else
    __builtin_amdgcn_tensor_load_to_lds(g0, g1, z4, z4, 0);
#endif
  }
  // TDM writes LDS outside the compiler's view: force it to treat sa_base's
  // memory as modified (otherwise s_a loads fold to undef and get hoisted).
  asm volatile("" : "+v"(sa_base) : : "memory");
#else
  // Cooperative fallback: 256 threads x 16 bytes.
  int row = tid >> 2, ch = (tid & 3) * 8;
  const u16t* src = A + ((long long)mblk + row) * K + k + ch;
  u16t* dst = sa_base + row * 32 + ch;
#pragma unroll
  for (int i = 0; i < 8; ++i) dst[i] = src[i];
#endif
}

// ---------------------------------------------------------------------------
// Fast GEMM: out[M,N](bf16) = A[M,K] * W[N,K]^T + bias1 + bias2, optional relu.
// Requires M%64==0, N%256==0, K%32==0. Block: 8 waves, tile 64(M) x 256(N);
// wave tile 32x64 (8 WMMA / K-step). A tile TDM double-buffered in LDS.
// ---------------------------------------------------------------------------
__global__ __launch_bounds__(256) void bilstm_k_gemm_fast(
    const u16t* __restrict__ A, const u16t* __restrict__ W,
    const float* __restrict__ bias1, const float* __restrict__ bias2,
    u16t* __restrict__ outp, int relu,
    int M, int N, int K, long long ldo, long long coff) {
  __shared__ u16t s_a[2][64][32];
  int tid = threadIdx.x, lane = tid & 31, wv = tid >> 5;
  int m0   = (wv >> 2) * 32;              // M offset within block tile
  int n0   = blockIdx.x * 256 + (wv & 3) * 64;
  int mblk = blockIdx.y * 64;

  bilstm_stage_a(A, M, K, mblk, 0, &s_a[0][0][0], 0u, tid);

  v8f acc[2][4] = {};
  int buf = 0;
  for (int k = 0; k < K; k += 32, buf ^= 1) {
    // Always keep one tile in flight; final iteration re-stages the same tile
    // into the dead buffer (drained by S_ENDPGM's implicit wait-idle).
    int kn = (k + 32 < K) ? (k + 32) : k;
    bilstm_stage_a(A, M, K, mblk, kn, &s_a[buf ^ 1][0][0],
                   (buf ^ 1) ? 4096u : 0u, tid);
    BILSTM_WAIT_TENSOR(1);   // current tile complete, next in flight
    __syncthreads();
    if (k + 32 < K)
      __builtin_prefetch(W + (long long)(n0 + (lane & 15)) * K + k + 32, 0, 3);

    v16bf a0 = load_frag_nc(&s_a[buf][0][0], 32, m0,      0, lane);
    v16bf a1 = load_frag_nc(&s_a[buf][0][0], 32, m0 + 16, 0, lane);
    v16bf b[4];
#pragma unroll
    for (int nj = 0; nj < 4; ++nj)
      b[nj] = load_frag_nc(W, K, n0 + nj * 16, k, lane);
#pragma unroll
    for (int nj = 0; nj < 4; ++nj) {
      acc[0][nj] = wmma_bf16(a0, b[nj], acc[0][nj]);
      acc[1][nj] = wmma_bf16_rb(a1, b[nj], acc[1][nj]);  // B cached from prev
    }
    __syncthreads();   // all reads of s_a[buf] done before it is re-staged
  }

  int cl = lane & 15, rb = (lane >> 4) * 8;
#pragma unroll
  for (int mi = 0; mi < 2; ++mi) {
#pragma unroll
    for (int nj = 0; nj < 4; ++nj) {
      int col = n0 + nj * 16 + cl;
      float bsum = (bias1 ? bias1[col] : 0.f) + (bias2 ? bias2[col] : 0.f);
#pragma unroll
      for (int r = 0; r < 8; ++r) {
        int row = mblk + m0 + mi * 16 + rb + r;
        float v = acc[mi][nj][r] + bsum;
        if (relu) v = fmaxf(v, 0.f);
        outp[(long long)row * ldo + coff + col] = f2bf(v);
      }
    }
  }
}

// ---------------------------------------------------------------------------
// Edge GEMM with bounds checks (used only for the ragged N=16/nstore=15 tail).
// ---------------------------------------------------------------------------
__global__ __launch_bounds__(256) void bilstm_k_gemm_edge(
    const u16t* __restrict__ A, const u16t* __restrict__ W,
    const float* __restrict__ bias1, const float* __restrict__ bias2,
    void* __restrict__ outp, int out_f32, int relu,
    int M, int N, int K, long long ldo, long long coff, int nstore) {
  int lane = threadIdx.x & 31, wv = threadIdx.x >> 5;
  int m0 = blockIdx.y * 64 + (wv >> 2) * 32;
  int n0 = blockIdx.x * 128 + (wv & 3) * 32;

  v8f acc[2][2] = {};
  for (int k = 0; k < K; k += 32) {
    v16bf a0 = load_frag_b(A, K, m0,      k, lane, M);
    v16bf a1 = load_frag_b(A, K, m0 + 16, k, lane, M);
    v16bf b0 = load_frag_b(W, K, n0,      k, lane, N);
    v16bf b1 = load_frag_b(W, K, n0 + 16, k, lane, N);
    acc[0][0] = wmma_bf16(a0, b0, acc[0][0]);
    acc[0][1] = wmma_bf16(a0, b1, acc[0][1]);
    acc[1][0] = wmma_bf16(a1, b0, acc[1][0]);
    acc[1][1] = wmma_bf16(a1, b1, acc[1][1]);
  }

  int cl = lane & 15, rb = (lane >> 4) * 8;
#pragma unroll
  for (int mi = 0; mi < 2; ++mi) {
#pragma unroll
    for (int ni = 0; ni < 2; ++ni) {
      int col = n0 + ni * 16 + cl;
      if (col >= nstore) continue;
      float bsum = (bias1 ? bias1[col] : 0.f) + (bias2 ? bias2[col] : 0.f);
#pragma unroll
      for (int r = 0; r < 8; ++r) {
        int row = m0 + mi * 16 + rb + r;
        if (row >= M) continue;
        float v = acc[mi][ni][r] + bsum;
        if (relu) v = fmaxf(v, 0.f);
        long long oi = (long long)row * ldo + coff + col;
        if (out_f32) ((float*)outp)[oi] = v;
        else         ((u16t*)outp)[oi]  = f2bf(v);
      }
    }
  }
}

// ---------------------------------------------------------------------------
// LSTM recurrent scan (one direction). Grid.x = B/16 batch tiles; 8 waves.
// Wave w owns hidden slice [32w, 32w+32): gate cols {g*256 + 32w .. +32}.
// h double-buffered in LDS (bf16); c lives in WMMA-accumulator-shaped VGPRs.
// All 8 WMMAs of a K-step share the A (h) fragment -> reuse-A hints.
// ---------------------------------------------------------------------------
__global__ __launch_bounds__(256) void bilstm_k_scan(
    const u16t* __restrict__ xp,    // [B*T, 1024] bf16, row = b*T + t
    const u16t* __restrict__ Whh,   // [1024, 256] bf16
    u16t* __restrict__ out,         // [B*T, ldo] bf16, h at col_off
    int T, long long ldo, long long col_off, int reverse) {
  const int H = 256, HS = H + 8;
  __shared__ u16t hbuf[2][16][H + 8];
  int lane = threadIdx.x & 31, wv = threadIdx.x >> 5;
  int b0 = blockIdx.x * 16;
  int colbase = wv * 32;

  for (int i = threadIdx.x; i < 16 * HS; i += 256) (&hbuf[0][0][0])[i] = 0;
  __syncthreads();

  v8f cst[2] = {};
  int cl = lane & 15, rb = (lane >> 4) * 8;

  for (int t = 0; t < T; ++t) {
    int tt  = reverse ? (T - 1 - t) : t;
    int cur = t & 1, nxt = cur ^ 1;

    v8f acc[4][2] = {};
    for (int k = 0; k < H; k += 32) {
      v16bf a = load_frag_nc(&hbuf[cur][0][0], HS, 0, k, lane);
#pragma unroll
      for (int g = 0; g < 4; ++g) {
        v16bf b0 = load_frag_nc(Whh, H, g * 256 + colbase,      k, lane);
        v16bf b1 = load_frag_nc(Whh, H, g * 256 + colbase + 16, k, lane);
        acc[g][0] = (g == 0) ? wmma_bf16(a, b0, acc[g][0])
                             : wmma_bf16_ra(a, b0, acc[g][0]);
        acc[g][1] = wmma_bf16_ra(a, b1, acc[g][1]);
      }
    }

#pragma unroll
    for (int ct = 0; ct < 2; ++ct) {
      int hcol = colbase + ct * 16 + cl;
#pragma unroll
      for (int r = 0; r < 8; ++r) {
        int row = rb + r;  // batch row in tile
        long long xrow = (long long)(b0 + row) * T + tt;
        const u16t* xr = xp + xrow * 1024;
        float gi = acc[0][ct][r] + bf2f(xr[hcol]);
        float gf = acc[1][ct][r] + bf2f(xr[256 + hcol]);
        float gg = acc[2][ct][r] + bf2f(xr[512 + hcol]);
        float go = acc[3][ct][r] + bf2f(xr[768 + hcol]);
        float si = 1.f / (1.f + __expf(-gi));
        float sf = 1.f / (1.f + __expf(-gf));
        float so = 1.f / (1.f + __expf(-go));
        float cn = sf * cst[ct][r] + si * tanhf(gg);
        cst[ct][r] = cn;
        u16t hb = f2bf(so * tanhf(cn));
        hbuf[nxt][row][hcol] = hb;
        out[xrow * ldo + col_off + hcol] = hb;
      }
    }
    __syncthreads();
  }
}

// ---------------------------------------------------------------------------
// Utility kernels
// ---------------------------------------------------------------------------
__global__ void bilstm_k_cast(const float* __restrict__ s, u16t* __restrict__ d,
                              long long n) {
  long long i = (long long)blockIdx.x * blockDim.x + threadIdx.x;
  if (i < n) d[i] = f2bf(s[i]);
}
__global__ void bilstm_k_zero16(u16t* __restrict__ d, long long n) {
  long long i = (long long)blockIdx.x * blockDim.x + threadIdx.x;
  if (i < n) d[i] = 0;
}
__global__ void bilstm_k_embed(const int* __restrict__ ids,
                               const float* __restrict__ tab,
                               u16t* __restrict__ d, int total) {
  int i = blockIdx.x * blockDim.x + threadIdx.x;
  if (i < total) {
    int tok = i >> 7, e = i & 127;
    d[i] = f2bf(tab[ids[tok] * 128 + e]);
  }
}
__global__ __launch_bounds__(128) void bilstm_k_gather(
    const int* __restrict__ wb, const u16t* __restrict__ wordOut,
    u16t* __restrict__ comb) {
  int bl = blockIdx.x;            // b*512 + l
  int b = bl >> 9, l = bl & 511;
  __shared__ int s_wid;
  if (threadIdx.x == 0) {
    const int* p = wb + b * 96;
    int cs = 0, wid = 0;
    for (int j = 0; j < 96; ++j) { cs += p[j]; wid += (cs <= l) ? 1 : 0; }
    s_wid = wid;
  }
  __syncthreads();
  int wid = s_wid;
  bool valid = wid < 96;
  int idx = valid ? wid : 95;
  const u16t* src = wordOut + ((long long)b * 96 + idx) * 512;
  u16t* dst = comb + (long long)bl * 1024 + 512;
  for (int c = threadIdx.x; c < 512; c += 128)
    dst[c] = valid ? src[c] : (u16t)0;
}

// ---------------------------------------------------------------------------
extern "C" void kernel_launch(void* const* d_in, const int* in_sizes, int n_in,
                              void* d_out, int out_size, void* d_ws,
                              size_t ws_size, hipStream_t stream) {
  (void)in_sizes; (void)n_in; (void)out_size; (void)ws_size;
  const int B = 64, L = 512, Wd = 96;
  const int ML = B * L;      // 32768
  const int MW = B * Wd;     // 6144

  const int*   char_ids = (const int*)  d_in[0];
  const float* word_emb = (const float*)d_in[1];
  const int*   word_bd  = (const int*)  d_in[2];
  const float* emb_tab  = (const float*)d_in[3];
  auto cpar = [&](int l, int d, int i) { return (const float*)d_in[4  + l * 8 + d * 4 + i]; };
  auto wpar = [&](int l, int d, int i) { return (const float*)d_in[20 + l * 8 + d * 4 + i]; };
  const float* cls_W1 = (const float*)d_in[36];
  const float* cls_b1 = (const float*)d_in[37];
  const float* cls_W2 = (const float*)d_in[38];
  const float* cls_b2 = (const float*)d_in[39];

  char* ws = (char*)d_ws;
  size_t off = 0;
  auto alloc16 = [&](size_t elems) -> u16t* {
    u16t* p = (u16t*)(ws + off);
    off += ((elems * 2) + 255) & ~(size_t)255;
    return p;
  };

  u16t* charX    = alloc16((size_t)ML * 128);
  u16t* wordX    = alloc16((size_t)MW * 768);
  u16t* xp       = alloc16((size_t)ML * 1024);
  u16t* charL1   = alloc16((size_t)ML * 512);
  u16t* combined = alloc16((size_t)ML * 1024);
  u16t* wordL1   = alloc16((size_t)MW * 512);
  u16t* wordOut  = alloc16((size_t)MW * 512);
  u16t* h1       = alloc16((size_t)ML * 512);
  u16t *cWih[2][2], *cWhh[2][2], *wWih[2][2], *wWhh[2][2];
  for (int l = 0; l < 2; ++l)
    for (int d = 0; d < 2; ++d) {
      cWih[l][d] = alloc16((size_t)1024 * (l == 0 ? 128 : 512));
      cWhh[l][d] = alloc16((size_t)1024 * 256);
      wWih[l][d] = alloc16((size_t)1024 * (l == 0 ? 768 : 512));
      wWhh[l][d] = alloc16((size_t)1024 * 256);
    }
  u16t* W1b = alloc16((size_t)512 * 1024);
  u16t* W2b = alloc16((size_t)16 * 512);   // row 15 zero-padded

  auto cast = [&](const float* s, u16t* d, long long n) {
    bilstm_k_cast<<<dim3((unsigned)((n + 255) / 256)), dim3(256), 0, stream>>>(s, d, n);
  };
  auto gemmF = [&](const u16t* A, const u16t* W, const float* b1, const float* b2,
                   u16t* out, int relu, int M, int N, int K,
                   long long ldo, long long coff) {
    dim3 g(N / 256, M / 64);
    bilstm_k_gemm_fast<<<g, dim3(256), 0, stream>>>(A, W, b1, b2, out, relu,
                                                    M, N, K, ldo, coff);
  };
  auto scan = [&](const u16t* xpb, const u16t* Whh, u16t* out, int T,
                  long long ldo, long long coff, int rev) {
    bilstm_k_scan<<<dim3(B / 16), dim3(256), 0, stream>>>(xpb, Whh, out, T, ldo, coff, rev);
  };

  // --- stage inputs / weights as bf16 ---
  cast(word_emb, wordX, (long long)MW * 768);
  bilstm_k_embed<<<dim3((ML * 128) / 256), dim3(256), 0, stream>>>(char_ids, emb_tab, charX, ML * 128);
  for (int l = 0; l < 2; ++l)
    for (int d = 0; d < 2; ++d) {
      cast(cpar(l, d, 0), cWih[l][d], (long long)1024 * (l == 0 ? 128 : 512));
      cast(cpar(l, d, 1), cWhh[l][d], (long long)1024 * 256);
      cast(wpar(l, d, 0), wWih[l][d], (long long)1024 * (l == 0 ? 768 : 512));
      cast(wpar(l, d, 1), wWhh[l][d], (long long)1024 * 256);
    }
  cast(cls_W1, W1b, (long long)512 * 1024);
  cast(cls_W2, W2b, (long long)15 * 512);
  bilstm_k_zero16<<<dim3(2), dim3(256), 0, stream>>>(W2b + (size_t)15 * 512, 512);

  // --- char pathway ---
  for (int d = 0; d < 2; ++d) {  // layer 0
    gemmF(charX, cWih[0][d], cpar(0, d, 2), cpar(0, d, 3), xp, 0,
          ML, 1024, 128, 1024, 0);
    scan(xp, cWhh[0][d], charL1, L, 512, d * 256, d);
  }
  for (int d = 0; d < 2; ++d) {  // layer 1 -> combined[:, 0:512]
    gemmF(charL1, cWih[1][d], cpar(1, d, 2), cpar(1, d, 3), xp, 0,
          ML, 1024, 512, 1024, 0);
    scan(xp, cWhh[1][d], combined, L, 1024, d * 256, d);
  }

  // --- word pathway ---
  for (int d = 0; d < 2; ++d) {  // layer 0
    gemmF(wordX, wWih[0][d], wpar(0, d, 2), wpar(0, d, 3), xp, 0,
          MW, 1024, 768, 1024, 0);
    scan(xp, wWhh[0][d], wordL1, Wd, 512, d * 256, d);
  }
  for (int d = 0; d < 2; ++d) {  // layer 1
    gemmF(wordL1, wWih[1][d], wpar(1, d, 2), wpar(1, d, 3), xp, 0,
          MW, 1024, 512, 1024, 0);
    scan(xp, wWhh[1][d], wordOut, Wd, 512, d * 256, d);
  }

  // --- expand word states to char positions -> combined[:, 512:1024] ---
  bilstm_k_gather<<<dim3(ML), dim3(128), 0, stream>>>(word_bd, wordOut, combined);

  // --- classifier ---
  gemmF(combined, W1b, cls_b1, nullptr, h1, 1, ML, 512, 1024, 512, 0);
  bilstm_k_gemm_edge<<<dim3(1, ML / 64), dim3(256), 0, stream>>>(
      h1, W2b, cls_b2, nullptr, d_out, 1, 0, ML, 16, 512, 15, 0, 15);
}
